// HybridJediBlock_52793738002858
// MI455X (gfx1250) — matrix operand order repaired and checked by reading, not compiled, not run on hardware
//
#include <hip/hip_runtime.h>

typedef __bf16 bf16;
typedef __attribute__((ext_vector_type(16))) __bf16 v16bf;
typedef __attribute__((ext_vector_type(8)))  __bf16 bf8v;
typedef __attribute__((ext_vector_type(8)))  float  v8f;

#define LDK 40   // padded LDS row stride in halves (80B -> conflict-free 16B frag loads)

enum { EP_NONE = 0, EP_BIAS = 1, EP_SCALE = 2, EP_SEL = 3, EP_SOFTPLUS = 4 };

// ---------------------------------------------------------------- utilities
__device__ __forceinline__ float blk_reduce(float v, float* sm, int tid, int n) {
  sm[tid] = v; __syncthreads();
  for (int s = n >> 1; s > 0; s >>= 1) { if (tid < s) sm[tid] += sm[tid + s]; __syncthreads(); }
  float r = sm[0]; __syncthreads();
  return r;
}

__device__ __forceinline__ float sigmoidf_(float x) { return 1.f / (1.f + expf(-x)); }

// gfx1250 async global->LDS copy (16B per lane), tracked with ASYNCcnt
__device__ __forceinline__ void async_ld16(unsigned int lds_addr, const bf16* gaddr) {
  asm volatile("global_load_async_to_lds_b128 %0, %1, off"
               :: "v"(lds_addr), "v"(gaddr) : "memory");
}
__device__ __forceinline__ void wait_async0() {
  asm volatile("s_wait_asynccnt 0x0" ::: "memory");
}
__device__ __forceinline__ unsigned int lds_addr_of(const void* p) {
  return (unsigned int)(uintptr_t)p;   // LDS aperture: low 32 bits = LDS byte offset
}

// ---------------------------------------------------------------- WMMA GEMM
// A-fragment (16x32, M rows): lane<16 -> K{koff..koff+7, koff+16..koff+23}, koff = (lane>>4)*8
__device__ __forceinline__ v16bf fragA_ld(const bf16* T, int r0, int lane) {
  const bf16* p = T + (r0 + (lane & 15)) * LDK + ((lane >> 4) << 3);
  bf8v lo = *(const bf8v*)p;
  bf8v hi = *(const bf8v*)(p + 16);
  return __builtin_shufflevector(lo, hi, 0,1,2,3,4,5,6,7,8,9,10,11,12,13,14,15);
}
// B-fragment (32x16 as W rows = columns of B): K{koff..koff+7, koff+16..koff+23}, koff = (lane>>4)*8
__device__ __forceinline__ v16bf fragB_ld(const bf16* T, int r0, int lane) {
  const bf16* p = T + (r0 + (lane & 15)) * LDK + ((lane >> 4) << 3);
  bf8v lo = *(const bf8v*)p;
  bf8v hi = *(const bf8v*)(p + 16);
  return __builtin_shufflevector(lo, hi, 0,1,2,3,4,5,6,7,8,9,10,11,12,13,14,15);
}

__device__ __forceinline__ void store_tile(v8f c, int m0, int n0, int lane,
                                           float* C, bf16* Cbf, int ldc, int ep,
                                           const float* bias, const float* g1,
                                           const float* g2, float alpha) {
  int n  = n0 + (lane & 15);
  int mb = m0 + ((lane >> 4) << 3);
  float bv = 0.f, gv = 0.f;
  if (ep == EP_BIAS || ep == EP_SEL || ep == EP_SOFTPLUS) bv = bias[n];
  if (ep == EP_SEL) gv = g1[n];
#pragma unroll
  for (int i = 0; i < 8; i++) {
    int m = mb + i;
    float y = c[i];
    if (ep == EP_BIAS)          y += bv;
    else if (ep == EP_SCALE)    y *= alpha;
    else if (ep == EP_SEL)      { float s = sigmoidf_((y + bv) * gv); y = g2[(size_t)m * ldc + n] * s; }
    else if (ep == EP_SOFTPLUS) { float t = y + bv; y = (t > 20.f) ? t : log1pf(expf(t)); }
    if (C)   C  [(size_t)m * ldc + n] = y;
    if (Cbf) Cbf[(size_t)m * ldc + n] = (bf16)y;
  }
}

// C[M,N] = A[M,K] (bf16) @ W[N,K]^T ; W fp32 (WBF=0, cvt in VGPRs) or bf16 (WBF=1, async to LDS)
// Double-buffered LDS pipeline: async copies into buf^1 overlap WMMA on buf.
template <int WBF>
__global__ __launch_bounds__(128) void gemm_wmma(
    const bf16* __restrict__ A, const void* __restrict__ Wp,
    float* __restrict__ C, bf16* __restrict__ Cbf,
    int K, int lda, int ldw, int ldc, int ep,
    const float* __restrict__ bias, const float* __restrict__ g1,
    const float* __restrict__ g2, float alpha) {
  __shared__ bf16 As[2][64 * LDK];
  __shared__ bf16 Ws[2][64 * LDK];
  int tid = threadIdx.x, lane = tid & 31, wid = tid >> 5;
  int bm = blockIdx.y * 64, bn = blockIdx.x * 64;
  int wm = (wid & 2) * 16, wn = (wid & 1) * 32;    // 2x2 wave grid -> 64x64 block tile
  v8f c00 = {}, c01 = {}, c10 = {}, c11 = {};
  int lr = tid >> 1, lc = (tid & 1) * 16;          // 64 rows, 2 threads/row, 16 elems each

  const bf16*  Ag  = A + (size_t)(bm + lr) * lda + lc;
  const bf16*  Wgb = (const bf16*) Wp + (size_t)(bn + lr) * ldw + lc;
  const float* Wgf = (const float*)Wp + (size_t)(bn + lr) * ldw + lc;

  unsigned int aL[2] = { lds_addr_of(&As[0][lr * LDK + lc]), lds_addr_of(&As[1][lr * LDK + lc]) };
  unsigned int wL[2] = { lds_addr_of(&Ws[0][lr * LDK + lc]), lds_addr_of(&Ws[1][lr * LDK + lc]) };

  // ---- prologue: stage k0 = 0 into buffer 0
  async_ld16(aL[0], Ag);
  async_ld16(aL[0] + 16, Ag + 8);
  if (WBF) {
    async_ld16(wL[0], Wgb);
    async_ld16(wL[0] + 16, Wgb + 8);
  } else {
    bf8v lo, hi;
#pragma unroll
    for (int i = 0; i < 8; i++) { lo[i] = (bf16)Wgf[i]; hi[i] = (bf16)Wgf[8 + i]; }
    bf16* dst = &Ws[0][lr * LDK + lc];
    *(bf8v*)dst = lo; *(bf8v*)(dst + 8) = hi;
  }
  wait_async0();
  __syncthreads();

  int buf = 0;
  for (int k0 = 0; k0 < K; k0 += 32) {
    int kn = k0 + 32;
    bool more = kn < K;
    float f[16];
    if (more) {
      // stage next tile into buf^1 while we compute on buf
      async_ld16(aL[buf ^ 1], Ag + kn);
      async_ld16(aL[buf ^ 1] + 16, Ag + kn + 8);
      if (WBF) {
        async_ld16(wL[buf ^ 1], Wgb + kn);
        async_ld16(wL[buf ^ 1] + 16, Wgb + kn + 8);
      } else {
#pragma unroll
        for (int i = 0; i < 16; i++) f[i] = Wgf[kn + i];
        __builtin_prefetch(Wgf + kn + 32, 0, 1);   // gfx1250 global_prefetch on the weight stream
      }
    }
    v16bf a0 = fragA_ld(&As[buf][0], wm, lane), a1 = fragA_ld(&As[buf][0], wm + 16, lane);
    v16bf b0 = fragB_ld(&Ws[buf][0], wn, lane), b1 = fragB_ld(&Ws[buf][0], wn + 16, lane);
    c00 = __builtin_amdgcn_wmma_f32_16x16x32_bf16(false, a0, false, b0, (short)0, c00, false, false);
    c01 = __builtin_amdgcn_wmma_f32_16x16x32_bf16(false, a0, false, b1, (short)0, c01, false, false);
    c10 = __builtin_amdgcn_wmma_f32_16x16x32_bf16(false, a1, false, b0, (short)0, c10, false, false);
    c11 = __builtin_amdgcn_wmma_f32_16x16x32_bf16(false, a1, false, b1, (short)0, c11, false, false);
    if (more && !WBF) {
      bf8v lo, hi;
#pragma unroll
      for (int i = 0; i < 8; i++) { lo[i] = (bf16)f[i]; hi[i] = (bf16)f[8 + i]; }
      bf16* dst = &Ws[buf ^ 1][lr * LDK + lc];
      *(bf8v*)dst = lo; *(bf8v*)(dst + 8) = hi;
    }
    wait_async0();
    __syncthreads();
    buf ^= 1;
  }
  store_tile(c00, bm + wm,      bn + wn,      lane, C, Cbf, ldc, ep, bias, g1, g2, alpha);
  store_tile(c01, bm + wm,      bn + wn + 16, lane, C, Cbf, ldc, ep, bias, g1, g2, alpha);
  store_tile(c10, bm + wm + 16, bn + wn,      lane, C, Cbf, ldc, ep, bias, g1, g2, alpha);
  store_tile(c11, bm + wm + 16, bn + wn + 16, lane, C, Cbf, ldc, ep, bias, g1, g2, alpha);
}

// ------------------------------------------------- stage kernels (f32 VALU)
__global__ __launch_bounds__(128) void k_norm_router(
    const float* __restrict__ x, const float* __restrict__ n1w,
    const float* __restrict__ rw, const float* __restrict__ snw,
    float* __restrict__ xn, bf16* __restrict__ xnbf,
    float* __restrict__ sn0, bf16* __restrict__ sn0bf, float* __restrict__ wts) {
  int l = blockIdx.x, t = threadIdx.x;
  __shared__ float red[128];
  float xv[8], ss = 0.f;
#pragma unroll
  for (int j = 0; j < 8; j++) { int d = t + 128 * j; xv[j] = x[l * 1024 + d]; ss += xv[j] * xv[j]; }
  ss = blk_reduce(ss, red, t, 128);
  float r1 = rsqrtf(ss * (1.f / 1024.f) + 1e-6f);
  float xnv[8], ss2 = 0.f, racc[4] = {0.f, 0.f, 0.f, 0.f};
#pragma unroll
  for (int j = 0; j < 8; j++) {
    int d = t + 128 * j;
    float v = xv[j] * r1 * n1w[d];
    xnv[j] = v; xn[l * 1024 + d] = v; xnbf[l * 1024 + d] = (bf16)v; ss2 += v * v;
#pragma unroll
    for (int e = 0; e < 4; e++) racc[e] += v * rw[e * 1024 + d];
  }
  ss2 = blk_reduce(ss2, red, t, 128);
  float r2 = rsqrtf(ss2 * (1.f / 1024.f) + 1e-6f);
#pragma unroll
  for (int j = 0; j < 8; j++) {
    int d = t + 128 * j;
    float v = xnv[j] * r2 * snw[d];
    sn0[l * 1024 + d] = v; sn0bf[l * 1024 + d] = (bf16)v;
  }
  float rl[4];
  for (int e = 0; e < 4; e++) rl[e] = blk_reduce(racc[e], red, t, 128);
  if (t == 0) {
    const float pri[4] = {0.5f, 0.2f, 0.15f, 0.15f};
    float m = fmaxf(fmaxf(rl[0], rl[1]), fmaxf(rl[2], rl[3]));
    float ev[4], s = 0.f;
    for (int e = 0; e < 4; e++) { ev[e] = expf(rl[e] - m); s += ev[e]; }
    float w[4], s2 = 0.f;
    for (int e = 0; e < 4; e++) { w[e] = ev[e] / s * pri[e]; s2 += w[e]; }
    for (int e = 0; e < 4; e++) wts[l * 4 + e] = w[e] / s2;
  }
}

__global__ void k_copy_dt(const float* __restrict__ proj, bf16* __restrict__ dtbf) {
  int i = blockIdx.x * 256 + threadIdx.x;     // 1024*64
  int m = i >> 6, j = i & 63;
  dtbf[i] = (bf16)proj[m * 320 + j];
}

// q head RMS + RoPE ; grid (H=16, L)
__global__ __launch_bounds__(64) void k_rope_q(
    const float* __restrict__ qlin, const float* __restrict__ qnw, bf16* __restrict__ qr) {
  int h = blockIdx.x, l = blockIdx.y, i = threadIdx.x;
  __shared__ float sm[64];
  __shared__ float sh[64];
  float v = qlin[l * 1024 + h * 64 + i];
  sm[i] = v * v; __syncthreads();
  for (int s = 32; s > 0; s >>= 1) { if (i < s) sm[i] += sm[i + s]; __syncthreads(); }
  float r = rsqrtf(sm[0] * (1.f / 64.f) + 1e-6f);
  float qn = v * r * qnw[i];
  sh[i] = qn; __syncthreads();
  int j = i & 31;
  float ang = (float)l * __powf(10000.f, -(float)j * (1.f / 32.f));
  float c = cosf(ang), s = sinf(ang);
  float other = (i < 32) ? -sh[i + 32] : sh[i - 32];
  qr[((size_t)h * 1024 + l) * 64 + i] = (bf16)(qn * c + other * s);
}

// k head RMS + RoPE + v transpose ; grid (KVH=8, L)
__global__ __launch_bounds__(64) void k_rope_kv(
    const float* __restrict__ klin, const float* __restrict__ vlin,
    const float* __restrict__ knw, bf16* __restrict__ kr, bf16* __restrict__ vT) {
  int h = blockIdx.x, l = blockIdx.y, i = threadIdx.x;
  __shared__ float sm[64];
  __shared__ float sh[64];
  float v = klin[l * 512 + h * 64 + i];
  sm[i] = v * v; __syncthreads();
  for (int s = 32; s > 0; s >>= 1) { if (i < s) sm[i] += sm[i + s]; __syncthreads(); }
  float r = rsqrtf(sm[0] * (1.f / 64.f) + 1e-6f);
  float kn = v * r * knw[i];
  sh[i] = kn; __syncthreads();
  int j = i & 31;
  float ang = (float)l * __powf(10000.f, -(float)j * (1.f / 32.f));
  float c = cosf(ang), s = sinf(ang);
  float other = (i < 32) ? -sh[i + 32] : sh[i - 32];
  kr[((size_t)h * 1024 + l) * 64 + i] = (bf16)(kn * c + other * s);
  vT[((size_t)h * 64 + i) * 1024 + l] = (bf16)vlin[l * 512 + h * 64 + i];
}

__global__ __launch_bounds__(256) void k_softmax_row(const float* __restrict__ S, bf16* __restrict__ P) {
  int row = blockIdx.x, t = threadIdx.x;
  __shared__ float red[256];
  float v[4], m = -1e30f;
#pragma unroll
  for (int j = 0; j < 4; j++) { v[j] = S[(size_t)row * 1024 + t + 256 * j]; m = fmaxf(m, v[j]); }
  red[t] = m; __syncthreads();
  for (int s = 128; s > 0; s >>= 1) { if (t < s) red[t] = fmaxf(red[t], red[t + s]); __syncthreads(); }
  m = red[0]; __syncthreads();
  float s = 0.f;
#pragma unroll
  for (int j = 0; j < 4; j++) { v[j] = expf(v[j] - m); s += v[j]; }
  s = blk_reduce(s, red, t, 256);
  float inv = 1.f / s;
#pragma unroll
  for (int j = 0; j < 4; j++) P[(size_t)row * 1024 + t + 256 * j] = (bf16)(v[j] * inv);
}

// SSM selective scan: one wave32 per channel d, 4 states/lane; grid 128 x 256
__global__ __launch_bounds__(256) void k_scan(
    const float* __restrict__ delta, const float* __restrict__ proj,
    const float* __restrict__ sn, const float* __restrict__ xn,
    const float* __restrict__ Alog, const float* __restrict__ Dp, bf16* __restrict__ ycomb) {
  int wid = threadIdx.x >> 5, lane = threadIdx.x & 31;
  int d = blockIdx.x * 8 + wid;
  const float* ar = Alog + (size_t)d * 128 + lane * 4;
  float a0 = -expf(ar[0]), a1 = -expf(ar[1]), a2 = -expf(ar[2]), a3 = -expf(ar[3]);
  float h0 = 0.f, h1 = 0.f, h2 = 0.f, h3 = 0.f;
  float dpd = Dp[d];
  for (int t = 0; t < 1024; t++) {
    float dt = delta[t * 1024 + d];
    float xt = sn[t * 1024 + d];
    float4 b  = *(const float4*)(proj + t * 320 + 64  + lane * 4);
    float4 cc = *(const float4*)(proj + t * 320 + 192 + lane * 4);
    float dx = dt * xt;
    h0 = __expf(dt * a0) * h0 + dx * b.x;
    h1 = __expf(dt * a1) * h1 + dx * b.y;
    h2 = __expf(dt * a2) * h2 + dx * b.z;
    h3 = __expf(dt * a3) * h3 + dx * b.w;
    float acc = h0 * cc.x + h1 * cc.y + h2 * cc.z + h3 * cc.w;
#pragma unroll
    for (int off = 16; off > 0; off >>= 1) acc += __shfl_xor(acc, off, 32);
    if (lane == 0) ycomb[t * 1024 + d] = (bf16)(acc + xn[t * 1024 + d] * dpd);
  }
}

// depthwise 3-tap conv + SiLU ; grid L x 256
__global__ __launch_bounds__(256) void k_conv(
    const float* __restrict__ xn, const float* __restrict__ dww,
    const float* __restrict__ dwb, bf16* __restrict__ convbf) {
  int l = blockIdx.x, t = threadIdx.x;
#pragma unroll
  for (int j = 0; j < 4; j++) {
    int d = t + 256 * j;
    float xm = (l > 0)    ? xn[(l - 1) * 1024 + d] : 0.f;
    float xc = xn[l * 1024 + d];
    float xp = (l < 1023) ? xn[(l + 1) * 1024 + d] : 0.f;
    float y = xm * dww[d * 3 + 0] + xc * dww[d * 3 + 1] + xp * dww[d * 3 + 2] + dwb[d];
    convbf[l * 1024 + d] = (bf16)(y * sigmoidf_(y));
  }
}

// branch mix + residual + norm2 + gate logits + top-2 ; one block per token
__global__ __launch_bounds__(128) void k_mix_gate(
    const float* __restrict__ x, const float* __restrict__ mem,
    const float* __restrict__ oss, const float* __restrict__ oat,
    const float* __restrict__ ocv, const float* __restrict__ wts,
    const float* __restrict__ n2w, const float* __restrict__ gw,
    float* __restrict__ x2, bf16* __restrict__ x2bf,
    float* __restrict__ vals0, float* __restrict__ vals1,
    int* __restrict__ idx0, int* __restrict__ idx1) {
  int l = blockIdx.x, t = threadIdx.x;
  __shared__ float red[128];
  float w0 = wts[l * 4], w1 = wts[l * 4 + 1], w2 = wts[l * 4 + 2], w3 = wts[l * 4 + 3];
  float xl[8], ss = 0.f;
#pragma unroll
  for (int j = 0; j < 8; j++) {
    int d = t + 128 * j; size_t o = (size_t)l * 1024 + d;
    float v = x[o] + w0 * oss[o] + w1 * oat[o] + w2 * ocv[o] + w3 * mem[o];
    xl[j] = v; x2[o] = v; x2bf[o] = (bf16)v; ss += v * v;
  }
  ss = blk_reduce(ss, red, t, 128);
  float r = rsqrtf(ss * (1.f / 1024.f) + 1e-6f);
  float g[8] = {0.f, 0.f, 0.f, 0.f, 0.f, 0.f, 0.f, 0.f};
#pragma unroll
  for (int j = 0; j < 8; j++) {
    int d = t + 128 * j;
    float xn2 = xl[j] * r * n2w[d];
#pragma unroll
    for (int e = 0; e < 8; e++) g[e] += xn2 * gw[e * 1024 + d];
  }
  for (int e = 0; e < 8; e++) g[e] = blk_reduce(g[e], red, t, 128);
  if (t == 0) {
    int i0 = 0; float v0 = g[0];
    for (int e = 1; e < 8; e++) if (g[e] > v0) { v0 = g[e]; i0 = e; }
    int i1 = -1; float v1 = -1e30f;
    for (int e = 0; e < 8; e++) if (e != i0 && g[e] > v1) { v1 = g[e]; i1 = e; }
    vals0[l] = v0; vals1[l] = v1; idx0[l] = i0; idx1[l] = i1;
  }
}

// softmax over sequence axis of top-k values (reference semantics)
__global__ __launch_bounds__(1024) void k_seq_softmax(
    const float* __restrict__ vals0, const float* __restrict__ vals1,
    float* __restrict__ gw0, float* __restrict__ gw1) {
  const float* src = blockIdx.x ? vals1 : vals0;
  float* dst = blockIdx.x ? gw1 : gw0;
  __shared__ float red[1024];
  int t = threadIdx.x;
  float v = src[t];
  red[t] = v; __syncthreads();
  for (int s = 512; s > 0; s >>= 1) { if (t < s) red[t] = fmaxf(red[t], red[t + s]); __syncthreads(); }
  float m = red[0]; __syncthreads();
  float e = expf(v - m);
  red[t] = e; __syncthreads();
  for (int s = 512; s > 0; s >>= 1) { if (t < s) red[t] += red[t + s]; __syncthreads(); }
  dst[t] = e / red[0];
}

__global__ void k_silugate(const float* __restrict__ h1, bf16* __restrict__ hh) {
  int i = blockIdx.x * 256 + threadIdx.x;   // 1024*4096
  int m = i >> 12, j = i & 4095;
  float a = h1[(size_t)m * 8192 + j];
  float g = h1[(size_t)m * 8192 + 4096 + j];
  hh[i] = (bf16)(a * sigmoidf_(a) * g);
}

__global__ __launch_bounds__(256) void k_final(
    const float* __restrict__ x2, const float* __restrict__ eo,
    const float* __restrict__ gw0, const float* __restrict__ gw1,
    const int* __restrict__ idx0, const int* __restrict__ idx1, float* __restrict__ out) {
  int l = blockIdx.x, t = threadIdx.x;
  int i0 = idx0[l], i1 = idx1[l];
  float w0 = gw0[l], w1 = gw1[l];
#pragma unroll
  for (int j = 0; j < 4; j++) {
    int d = t + 256 * j;
    float m = w0 * eo[((size_t)i0 * 1024 + l) * 1024 + d] +
              w1 * eo[((size_t)i1 * 1024 + l) * 1024 + d];
    out[(size_t)l * 1024 + d] = x2[(size_t)l * 1024 + d] + 0.1f * m;
  }
}

// ---------------------------------------------------------------- launcher
extern "C" void kernel_launch(void* const* d_in, const int* in_sizes, int n_in,
                              void* d_out, int out_size, void* d_ws, size_t ws_size,
                              hipStream_t stream) {
  const float* x        = (const float*)d_in[0];
  const float* mem_out  = (const float*)d_in[1];
  const float* norm1_w  = (const float*)d_in[2];
  const float* router_w = (const float*)d_in[3];
  const float* ssm_nw   = (const float*)d_in[4];
  const float* x_proj_w = (const float*)d_in[5];
  const float* dt_proj_w= (const float*)d_in[6];
  const float* dt_proj_b= (const float*)d_in[7];
  const float* A_log    = (const float*)d_in[8];
  const float* D_param  = (const float*)d_in[9];
  const float* ssm_out_w= (const float*)d_in[10];
  const float* sel_w    = (const float*)d_in[11];
  const float* sel_b    = (const float*)d_in[12];
  const float* sel_gate = (const float*)d_in[13];
  const float* q_w      = (const float*)d_in[14];
  const float* k_w      = (const float*)d_in[15];
  const float* v_w      = (const float*)d_in[16];
  const float* o_w      = (const float*)d_in[17];
  const float* qn_w     = (const float*)d_in[18];
  const float* kn_w     = (const float*)d_in[19];
  const float* dw_w     = (const float*)d_in[20];
  const float* dw_b     = (const float*)d_in[21];
  const float* pw_w     = (const float*)d_in[22];
  const float* pw_b     = (const float*)d_in[23];
  const float* norm2_w  = (const float*)d_in[24];
  const float* gate_w   = (const float*)d_in[25];
  const float* e_w1     = (const float*)d_in[26];
  const float* e_w2     = (const float*)d_in[27];
  const float* e_lw     = (const float*)d_in[28];
  const float* e_lb     = (const float*)d_in[29];
  float* out = (float*)d_out;

  const int L = 1024, H = 16, KVH = 8;
  const size_t LD = (size_t)L * 1024;

  size_t off = 0; char* base = (char*)d_ws;
  auto alloc = [&](size_t bytes) -> void* {
    void* r = base + off; off = (off + bytes + 255) & ~(size_t)255; return r;
  };
  float* xn     = (float*)alloc(LD * 4);
  bf16*  xnbf   = (bf16*) alloc(LD * 2);
  float* sn0    = (float*)alloc(LD * 4);
  bf16*  sn0bf  = (bf16*) alloc(LD * 2);
  float* sn     = (float*)alloc(LD * 4);
  bf16*  snbf   = (bf16*) alloc(LD * 2);
  float* proj   = (float*)alloc((size_t)L * 320 * 4);
  bf16*  dtbf   = (bf16*) alloc((size_t)L * 64 * 2);
  float* delta  = (float*)alloc(LD * 4);
  float* qlin   = (float*)alloc(LD * 4);
  float* klin   = (float*)alloc((size_t)L * 512 * 4);
  float* vlin   = (float*)alloc((size_t)L * 512 * 4);
  bf16*  qr     = (bf16*) alloc(LD * 2);
  bf16*  kr     = (bf16*) alloc((size_t)L * 512 * 2);
  bf16*  vT     = (bf16*) alloc((size_t)L * 512 * 2);
  float* S      = (float*)alloc((size_t)L * L * 4);
  bf16*  P      = (bf16*) alloc((size_t)L * L * 2);
  float* attn   = (float*)alloc(LD * 4);
  bf16*  attnbf = (bf16*) alloc(LD * 2);
  float* out_at = (float*)alloc(LD * 4);
  bf16*  ycomb  = (bf16*) alloc(LD * 2);
  float* out_ss = (float*)alloc(LD * 4);
  bf16*  convbf = (bf16*) alloc(LD * 2);
  float* out_cv = (float*)alloc(LD * 4);
  float* wts    = (float*)alloc((size_t)L * 4 * 4);
  float* x2     = (float*)alloc(LD * 4);
  bf16*  x2bf   = (bf16*) alloc(LD * 2);
  float* vals0  = (float*)alloc(L * 4);
  float* vals1  = (float*)alloc(L * 4);
  int*   idx0   = (int*)  alloc(L * 4);
  int*   idx1   = (int*)  alloc(L * 4);
  float* gw0    = (float*)alloc(L * 4);
  float* gw1    = (float*)alloc(L * 4);
  float* h1     = (float*)alloc((size_t)L * 8192 * 4);
  bf16*  hhbf   = (bf16*) alloc((size_t)L * 4096 * 2);
  bf16*  t2bf   = (bf16*) alloc(LD * 2);
  float* eo     = (float*)alloc((size_t)8 * LD * 4);

  auto G = [](int N, int M) { return dim3((unsigned)(N / 64), (unsigned)(M / 64)); };

  // norm1 + router + ssm_norm
  k_norm_router<<<L, 128, 0, stream>>>(x, norm1_w, router_w, ssm_nw, xn, xnbf, sn0, sn0bf, wts);

  // sel = sigmoid((sn0 @ sel_w^T + b) * g); sn = sn0 * sel   (fused epilogue)
  gemm_wmma<0><<<G(1024, 1024), 128, 0, stream>>>(sn0bf, sel_w, sn, snbf,
      1024, 1024, 1024, 1024, EP_SEL, sel_b, sel_gate, sn0, 0.f);

  // proj = sn @ x_proj_w^T   (L x 320)
  gemm_wmma<0><<<G(320, 1024), 128, 0, stream>>>(snbf, x_proj_w, proj, nullptr,
      1024, 1024, 1024, 320, EP_NONE, nullptr, nullptr, nullptr, 0.f);
  k_copy_dt<<<(L * 64) / 256, 256, 0, stream>>>(proj, dtbf);

  // delta = softplus(proj_dt @ dt_proj_w^T + b)
  gemm_wmma<0><<<G(1024, 1024), 128, 0, stream>>>(dtbf, dt_proj_w, delta, nullptr,
      64, 64, 64, 1024, EP_SOFTPLUS, dt_proj_b, nullptr, nullptr, 0.f);

  // q/k/v projections
  gemm_wmma<0><<<G(1024, 1024), 128, 0, stream>>>(xnbf, q_w, qlin, nullptr,
      1024, 1024, 1024, 1024, EP_NONE, nullptr, nullptr, nullptr, 0.f);
  gemm_wmma<0><<<G(512, 1024), 128, 0, stream>>>(xnbf, k_w, klin, nullptr,
      1024, 1024, 1024, 512, EP_NONE, nullptr, nullptr, nullptr, 0.f);
  gemm_wmma<0><<<G(512, 1024), 128, 0, stream>>>(xnbf, v_w, vlin, nullptr,
      1024, 1024, 1024, 512, EP_NONE, nullptr, nullptr, nullptr, 0.f);

  k_rope_q <<<dim3(H, L),  64, 0, stream>>>(qlin, qn_w, qr);
  k_rope_kv<<<dim3(KVH, L), 64, 0, stream>>>(klin, vlin, kn_w, kr, vT);

  // SSM scan -> ycomb = y_ssm + xn*D
  k_scan<<<128, 256, 0, stream>>>(delta, proj, sn, xn, A_log, D_param, ycomb);
  gemm_wmma<0><<<G(1024, 1024), 128, 0, stream>>>(ycomb, ssm_out_w, out_ss, nullptr,
      1024, 1024, 1024, 1024, EP_NONE, nullptr, nullptr, nullptr, 0.f);

  // conv branch
  k_conv<<<L, 256, 0, stream>>>(xn, dw_w, dw_b, convbf);
  gemm_wmma<0><<<G(1024, 1024), 128, 0, stream>>>(convbf, pw_w, out_cv, nullptr,
      1024, 1024, 1024, 1024, EP_BIAS, pw_b, nullptr, nullptr, 0.f);

  // attention (per head; GQA: kv head = h/2) — both operands bf16, fully async tiles
  for (int h = 0; h < H; h++) {
    int kh = h >> 1;
    gemm_wmma<1><<<G(1024, 1024), 128, 0, stream>>>(
        qr + (size_t)h * L * 64, kr + (size_t)kh * L * 64, S, nullptr,
        64, 64, 64, 1024, EP_SCALE, nullptr, nullptr, nullptr, 0.125f);
    k_softmax_row<<<L, 256, 0, stream>>>(S, P);
    gemm_wmma<1><<<G(64, 1024), 128, 0, stream>>>(
        P, vT + (size_t)kh * 64 * L, attn + h * 64, attnbf + h * 64,
        1024, 1024, 1024, 1024, EP_NONE, nullptr, nullptr, nullptr, 0.f);
  }
  gemm_wmma<0><<<G(1024, 1024), 128, 0, stream>>>(attnbf, o_w, out_at, nullptr,
      1024, 1024, 1024, 1024, EP_NONE, nullptr, nullptr, nullptr, 0.f);

  // mix + residual + norm2 + gate top-2 ; sequence-axis softmax of top-k vals
  k_mix_gate<<<L, 128, 0, stream>>>(x, mem_out, out_ss, out_at, out_cv, wts,
      norm2_w, gate_w, x2, x2bf, vals0, vals1, idx0, idx1);
  k_seq_softmax<<<2, 1024, 0, stream>>>(vals0, vals1, gw0, gw1);

  // dense MoE (matches reference: all experts computed, top-2 combined)
  for (int e = 0; e < 8; e++) {
    gemm_wmma<0><<<G(8192, 1024), 128, 0, stream>>>(
        x2bf, e_w1 + (size_t)e * 8192 * 1024, h1, nullptr,
        1024, 1024, 1024, 8192, EP_NONE, nullptr, nullptr, nullptr, 0.f);
    k_silugate<<<(L * 4096) / 256, 256, 0, stream>>>(h1, hhbf);
    gemm_wmma<0><<<G(1024, 1024), 128, 0, stream>>>(
        hhbf, e_w2 + (size_t)e * 1024 * 4096, nullptr, t2bf,
        4096, 4096, 4096, 1024, EP_NONE, nullptr, nullptr, nullptr, 0.f);
    gemm_wmma<0><<<G(1024, 1024), 128, 0, stream>>>(
        t2bf, e_lw + (size_t)e * 1024 * 1024, eo + (size_t)e * LD, nullptr,
        1024, 1024, 1024, 1024, EP_BIAS, e_lb + e * 1024, nullptr, nullptr, 0.f);
  }

  k_final<<<L, 256, 0, stream>>>(x2, eo, gw0, gw1, idx0, idx1, out);
}
